// MultiheadSelfAttention_33672543600672
// MI455X (gfx1250) — matrix operand order repaired and checked by reading, not compiled, not run on hardware
//
#include <hip/hip_runtime.h>
#include <hip/hip_bf16.h>

// ---------------------------------------------------------------------------
// Multi-head self-attention for gfx1250 (MI455X), bf16 WMMA + fp32 accum.
//   x[2,2048,512] fp32 -> out[2,2048,512] fp32
// convert -> QKV proj (16x64 WMMA tiles) ->
//   flash attention (K via async global->LDS, Vt via TDM tensor_load_to_lds,
//                    double-buffered; WMMA scores/AV; LDS P transpose) ->
//   out proj (16x64 WMMA tiles)
// ---------------------------------------------------------------------------

typedef __attribute__((ext_vector_type(16))) __bf16 v16bf;
typedef __attribute__((ext_vector_type(8)))  __bf16 v8bf;
typedef __attribute__((ext_vector_type(8)))  float  v8f;
typedef __attribute__((ext_vector_type(4)))  unsigned int u32x4;
typedef __attribute__((ext_vector_type(4)))  int i32x4;
typedef __attribute__((ext_vector_type(8)))  int i32x8;

#define DM   512
#define SEQ  2048
#define BATCH 2
#define HEADS 8
#define HD   64
#define NTOK (BATCH*SEQ)          // 4096

__device__ __forceinline__ unsigned short f2bf(float f) {
  unsigned int u = __builtin_bit_cast(unsigned int, f);
  u += 0x7FFFu + ((u >> 16) & 1u);          // round-to-nearest-even
  return (unsigned short)(u >> 16);
}

union FragU { v16bf v; v8bf h[2]; };

// A-fragment (16x32 bf16). Lane L: row M = L%16; K chunks (L/16)*8..+7 and
// 16+(L/16)*8..+7 (ISA 05_wmma 16-bit A layout). Two 16B loads.
__device__ __forceinline__ v16bf load_a(const unsigned short* p, int row,
                                        int ldk, int k0, int lane) {
  FragU f;
  const unsigned short* q = p + row * ldk + k0 + ((lane >> 4) << 3);
  f.h[0] = *(const v8bf*)(q);
  f.h[1] = *(const v8bf*)(q + 16);
  return f.v;
}

// B-fragment (32x16 bf16). Lane L: column N = L%16; K chunks (L/16)*8..+7
// and 16+(L/16)*8..+7. Column N of B == row `col` of a row-major [N,ldk] matrix.
__device__ __forceinline__ v16bf load_b(const unsigned short* p, int col,
                                        int ldk, int k0, int lane) {
  FragU f;
  const unsigned short* q = p + col * ldk + k0 + ((lane >> 4) << 3);
  f.h[0] = *(const v8bf*)(q);
  f.h[1] = *(const v8bf*)(q + 16);
  return f.v;
}

__device__ __forceinline__ v8f wmma_bf16(v16bf a, v16bf b, v8f c) {
  return __builtin_amdgcn_wmma_f32_16x16x32_bf16(false, a, false, b,
                                                 (short)0, c, false, false);
}

// Async global->LDS 16B copy (ASYNCcnt-tracked, ISA 08_async_tensor).
__device__ __forceinline__ void async_b128(unsigned lds_off, const void* g) {
  unsigned long long ga = (unsigned long long)(__SIZE_TYPE__)g;
  asm volatile("global_load_async_to_lds_b128 %0, %1, off"
               :: "v"(lds_off), "v"(ga) : "memory");
}

// TDM: DMA a 2D tile (tile_dim0 x tile_dim1 elements of 2B, row stride
// `stride` elements) from global into LDS at byte offset lds_addr, packed
// contiguously (ISA 08_async_tensor D# groups). TENSORcnt-tracked.
__device__ __forceinline__ void tdm_load_2d_b16(unsigned lds_addr,
                                                const void* gaddr,
                                                unsigned tensor_d0,
                                                unsigned tensor_d1,
                                                unsigned tile_d0,
                                                unsigned tile_d1,
                                                unsigned stride) {
  unsigned long long ga = (unsigned long long)(__SIZE_TYPE__)gaddr;
  u32x4 g0 = { 1u,                                  // count=1, user descriptor
               lds_addr,                            // LDS byte address
               (unsigned)(ga & 0xFFFFFFFFull),      // global_addr[31:0]
               (unsigned)(((ga >> 32) & 0x01FFFFFFull) | (2u << 30)) }; // type=2
  i32x8 g1 = { (int)(1u << 16),                     // data_size=1 (2 bytes)
               (int)((tensor_d0 & 0xFFFFu) << 16),  // tensor_dim0[15:0]
               (int)((tensor_d0 >> 16) | ((tensor_d1 & 0xFFFFu) << 16)),
               (int)((tensor_d1 >> 16) | (tile_d0 << 16)),   // tile_dim0
               (int)(tile_d1),                      // tile_dim1 (tile_dim2=0)
               (int)(stride),                       // tensor_dim0_stride[31:0]
               0, 0 };
  i32x4 z4 = { 0, 0, 0, 0 };
#if defined(__clang_major__) && (__clang_major__ >= 23)
  i32x8 z8 = { 0, 0, 0, 0, 0, 0, 0, 0 };
  __builtin_amdgcn_tensor_load_to_lds(g0, g1, z4, z4, z8, 0);
#else
  __builtin_amdgcn_tensor_load_to_lds(g0, g1, z4, z4, 0);
#endif
}

// --------------------------- fp32 -> bf16 ----------------------------------
__global__ __launch_bounds__(256) void k_to_bf16(const float* __restrict__ in,
                                                 unsigned short* __restrict__ out,
                                                 int n) {
  int i = blockIdx.x * 256 + threadIdx.x;
  if (i < n) out[i] = f2bf(in[i]);
}

// --------------------------- QKV projection --------------------------------
// y = x @ W^T + bias (then *scale). One wave per 16x64 tile (A reused 4x).
// mode 0: Q -> [b,h,s,d]  (scale = log2(e)/sqrt(hd) folded in for exp2 softmax)
// mode 1: K -> [b,h,s,d]
// mode 2: V -> [b,h,d,s]  (transposed so AV B-frags are contiguous)
__global__ __launch_bounds__(256) void k_proj_qkv(
    const unsigned short* __restrict__ xb,
    const unsigned short* __restrict__ wb,
    const float* __restrict__ bias,
    unsigned short* __restrict__ out,
    int mode, float scale) {
  int lane = threadIdx.x & 31;
  int gw = blockIdx.x * 8 + (threadIdx.x >> 5);  // 2048 waves
  int tn = gw & 7;                  // 8 col-groups of 64
  int tm = gw >> 3;                 // 256 row tiles
  int arow = tm * 16 + (lane & 15);
  int bc = tn * 64 + (lane & 15);
  v8f acc[4] = {{}, {}, {}, {}};
  for (int k0 = 0; k0 < DM; k0 += 32) {
    v16bf a  = load_a(xb, arow, DM, k0, lane);
    v16bf b0 = load_b(wb, bc,      DM, k0, lane);
    v16bf b1 = load_b(wb, bc + 16, DM, k0, lane);
    v16bf b2 = load_b(wb, bc + 32, DM, k0, lane);
    v16bf b3 = load_b(wb, bc + 48, DM, k0, lane);
    acc[0] = wmma_bf16(a, b0, acc[0]);
    acc[1] = wmma_bf16(a, b1, acc[1]);
    acc[2] = wmma_bf16(a, b2, acc[2]);
    acc[3] = wmma_bf16(a, b3, acc[3]);
  }
#pragma unroll
  for (int s = 0; s < 4; ++s) {
    int n = tn * 64 + s * 16 + (lane & 15);
    float bv = bias[n];
    int h = n >> 6, d = n & 63;
#pragma unroll
    for (int i = 0; i < 8; ++i) {
      int m = tm * 16 + i + ((lane >> 4) << 3);   // C layout: M = i + 8*(L/16)
      int b_ = m >> 11, sq = m & 2047;
      unsigned short o = f2bf((acc[s][i] + bv) * scale);
      if (mode == 2)
        out[(((b_ * HEADS + h) * HD + d) * SEQ) + sq] = o;   // Vt [b,h,d,s]
      else
        out[(((b_ * HEADS + h) * SEQ + sq) * HD) + d] = o;   // [b,h,s,d]
    }
  }
}

// --------------------------- flash attention -------------------------------
// One block = 8 query-tiles of the same (b,h); 32 keys per step.
// K chunk (32x64 bf16, 4KB, contiguous): all 256 threads, one async b128 each.
// Vt chunk (64 rows x 32 keys, row stride 2048): single TDM descriptor issued
// by wave 0. Both double-buffered; counted s_wait_asynccnt / s_wait_tensorcnt
// overlap the next chunk's DMA with this chunk's 8 WMMAs + softmax VALU.
__global__ __launch_bounds__(256) void k_attn(
    const unsigned short* __restrict__ qb,
    const unsigned short* __restrict__ kb,
    const unsigned short* __restrict__ vtb,
    unsigned short* __restrict__ ctx) {
  __shared__ __align__(128) unsigned char smem[2 * 8192 + 8 * 1024]; // 24KB
  int tid = threadIdx.x;
  int lane = tid & 31;
  int w = tid >> 5;
  int bh = blockIdx.x >> 4;                 // 16 (b,h) pairs
  int qt = (blockIdx.x & 15) * 8 + w;       // 128 query tiles
  const unsigned short* Qp = qb + bh * SEQ * HD;
  const unsigned char* KpB = (const unsigned char*)(kb + bh * SEQ * HD);
  const unsigned char* VpB = (const unsigned char*)(vtb + bh * HD * SEQ);
  unsigned short* pbuf = (unsigned short*)&smem[16384 + w * 1024];

  int qrow = qt * 16 + (lane & 15);
  v16bf aq0 = load_a(Qp, qrow, HD, 0, lane);    // K = 0..31 of head dim
  v16bf aq1 = load_a(Qp, qrow, HD, 32, lane);   // K = 32..63

  float mrow[8], lrow[8];
  v8f o0 = {}, o1 = {}, o2 = {}, o3 = {};
#pragma unroll
  for (int i = 0; i < 8; ++i) { mrow[i] = -1e30f; lrow[i] = 0.f; }

  // K chunk: rows j0..j0+31 are contiguous 4KB; 16B per thread.
  auto stage_k = [&](int buf, int j0) {
    unsigned o = (unsigned)tid * 16u;                 // 0..4080
    unsigned dst = (unsigned)(__SIZE_TYPE__)&smem[buf * 8192 + o];
    async_b128(dst, KpB + (unsigned)(j0 * 128) + o);
  };
  // Vt chunk: 64 rows x 32 elements, row stride 2048 elements -> one TDM op.
  auto stage_v = [&](int buf, int j0) {
    unsigned dst = (unsigned)(__SIZE_TYPE__)&smem[buf * 8192 + 4096];
    tdm_load_2d_b16(dst, VpB + (unsigned)(j0 * 2),
                    /*tensor_d0=*/SEQ, /*tensor_d1=*/HD,
                    /*tile_d0=*/32, /*tile_d1=*/HD, /*stride=*/SEQ);
  };

  stage_k(0, 0);
  if (w == 0) stage_v(0, 0);
  for (int j0 = 0; j0 < SEQ; j0 += 32) {
    int c = (j0 >> 5) & 1;
    __syncthreads();                     // prior reads of buf[c^1] complete
    bool more = (j0 + 32) < SEQ;
    if (more) {
      stage_k(c ^ 1, j0 + 32);           // issue next chunk's DMA
      if (w == 0) stage_v(c ^ 1, j0 + 32);
    }
    if (more) asm volatile("s_wait_asynccnt 1" ::: "memory"); // cur K done
    else      asm volatile("s_wait_asynccnt 0" ::: "memory");
    if (w == 0) {                        // cur V TDM done (in-order per wave)
      if (more) __builtin_amdgcn_s_wait_tensorcnt(1);
      else      __builtin_amdgcn_s_wait_tensorcnt(0);
    }
    __syncthreads();                     // all staging visible block-wide
    const unsigned short* Ks = (const unsigned short*)&smem[c * 8192];
    const unsigned short* Vs = (const unsigned short*)&smem[c * 8192 + 4096];

    // scores: two 16x16 tiles covering keys j0..j0+31 (frags from LDS)
    v16bf bk00 = load_b(Ks, (lane & 15),      HD, 0,  lane);
    v16bf bk01 = load_b(Ks, (lane & 15),      HD, 32, lane);
    v16bf bk10 = load_b(Ks, 16 + (lane & 15), HD, 0,  lane);
    v16bf bk11 = load_b(Ks, 16 + (lane & 15), HD, 32, lane);
    v8f s0 = {}, s1 = {};
    s0 = wmma_bf16(aq0, bk00, s0);
    s0 = wmma_bf16(aq1, bk01, s0);
    s1 = wmma_bf16(aq0, bk10, s1);
    s1 = wmma_bf16(aq1, bk11, s1);

    // online softmax in base-2 (log2e folded into Q); rows span 16 lanes
#pragma unroll
    for (int i = 0; i < 8; ++i) {
      float mx = fmaxf(s0[i], s1[i]);
      for (int t = 1; t < 16; t <<= 1) mx = fmaxf(mx, __shfl_xor(mx, t, 32));
      float nm = fmaxf(mrow[i], mx);
      float sc = exp2f(mrow[i] - nm);
      float p0 = exp2f(s0[i] - nm);
      float p1 = exp2f(s1[i] - nm);
      float rs = p0 + p1;
      for (int t = 1; t < 16; t <<= 1) rs += __shfl_xor(rs, t, 32);
      lrow[i] = lrow[i] * sc + rs;
      mrow[i] = nm;
      o0[i] *= sc; o1[i] *= sc; o2[i] *= sc; o3[i] *= sc;
      int r = i + ((lane >> 4) << 3);
      pbuf[r * 32 + (lane & 15)]      = f2bf(p0);
      pbuf[r * 32 + 16 + (lane & 15)] = f2bf(p1);
    }

    // P as A-frag (16x32) from LDS; V B-frags from staged Vt chunk
    v16bf ap  = load_a(pbuf, lane & 15, 32, 0, lane);
    v16bf bv0 = load_b(Vs, 0  + (lane & 15), 32, 0, lane);
    v16bf bv1 = load_b(Vs, 16 + (lane & 15), 32, 0, lane);
    v16bf bv2 = load_b(Vs, 32 + (lane & 15), 32, 0, lane);
    v16bf bv3 = load_b(Vs, 48 + (lane & 15), 32, 0, lane);
    o0 = wmma_bf16(ap, bv0, o0);
    o1 = wmma_bf16(ap, bv1, o1);
    o2 = wmma_bf16(ap, bv2, o2);
    o3 = wmma_bf16(ap, bv3, o3);
  }

  // epilogue: O /= l, write ctx bf16 as [token, h*64+d] for the out-proj GEMM
  int b_ = bh >> 3, h = bh & 7;
#pragma unroll
  for (int i = 0; i < 8; ++i) {
    float inv = 1.0f / lrow[i];
    int srow = qt * 16 + i + ((lane >> 4) << 3);
    int token = b_ * SEQ + srow;
    int col = h * HD + (lane & 15);
    unsigned short* cp = ctx + token * DM + col;
    cp[0]  = f2bf(o0[i] * inv);
    cp[16] = f2bf(o1[i] * inv);
    cp[32] = f2bf(o2[i] * inv);
    cp[48] = f2bf(o3[i] * inv);
  }
}

// --------------------------- output projection -----------------------------
__global__ __launch_bounds__(256) void k_out_proj(
    const unsigned short* __restrict__ ctxb,
    const unsigned short* __restrict__ wob,
    const float* __restrict__ bias,
    float* __restrict__ out) {
  int lane = threadIdx.x & 31;
  int gw = blockIdx.x * 8 + (threadIdx.x >> 5);  // 2048 waves
  int tn = gw & 7;
  int tm = gw >> 3;
  int arow = tm * 16 + (lane & 15);
  int bc = tn * 64 + (lane & 15);
  v8f acc[4] = {{}, {}, {}, {}};
  for (int k0 = 0; k0 < DM; k0 += 32) {
    v16bf a  = load_a(ctxb, arow, DM, k0, lane);
    v16bf b0 = load_b(wob, bc,      DM, k0, lane);
    v16bf b1 = load_b(wob, bc + 16, DM, k0, lane);
    v16bf b2 = load_b(wob, bc + 32, DM, k0, lane);
    v16bf b3 = load_b(wob, bc + 48, DM, k0, lane);
    acc[0] = wmma_bf16(a, b0, acc[0]);
    acc[1] = wmma_bf16(a, b1, acc[1]);
    acc[2] = wmma_bf16(a, b2, acc[2]);
    acc[3] = wmma_bf16(a, b3, acc[3]);
  }
#pragma unroll
  for (int s = 0; s < 4; ++s) {
    int n = tn * 64 + s * 16 + (lane & 15);
    float bv = bias[n];
#pragma unroll
    for (int i = 0; i < 8; ++i) {
      int m = tm * 16 + i + ((lane >> 4) << 3);
      out[m * DM + n] = acc[s][i] + bv;
    }
  }
}

// ---------------------------------------------------------------------------
extern "C" void kernel_launch(void* const* d_in, const int* in_sizes, int n_in,
                              void* d_out, int out_size, void* d_ws, size_t ws_size,
                              hipStream_t stream) {
  const float* x  = (const float*)d_in[0];
  const float* Wq = (const float*)d_in[1];
  const float* bq = (const float*)d_in[2];
  const float* Wk = (const float*)d_in[3];
  const float* bk = (const float*)d_in[4];
  const float* Wv = (const float*)d_in[5];
  const float* bv = (const float*)d_in[6];
  const float* Wo = (const float*)d_in[7];
  const float* bo = (const float*)d_in[8];
  float* out = (float*)d_out;

  const int X_ELEMS = NTOK * DM;      // 2,097,152
  const int W_ELEMS = DM * DM;        //   262,144

  unsigned short* base = (unsigned short*)d_ws;
  unsigned short* xb  = base;
  unsigned short* wqb = xb  + X_ELEMS;
  unsigned short* wkb = wqb + W_ELEMS;
  unsigned short* wvb = wkb + W_ELEMS;
  unsigned short* wob = wvb + W_ELEMS;
  unsigned short* qbf = wob + W_ELEMS;
  unsigned short* kbf = qbf + X_ELEMS;
  unsigned short* vtb = kbf + X_ELEMS;
  unsigned short* ctx = vtb + X_ELEMS;

  // 1) fp32 -> bf16 staging
  k_to_bf16<<<X_ELEMS / 256, 256, 0, stream>>>(x,  xb,  X_ELEMS);
  k_to_bf16<<<W_ELEMS / 256, 256, 0, stream>>>(Wq, wqb, W_ELEMS);
  k_to_bf16<<<W_ELEMS / 256, 256, 0, stream>>>(Wk, wkb, W_ELEMS);
  k_to_bf16<<<W_ELEMS / 256, 256, 0, stream>>>(Wv, wvb, W_ELEMS);
  k_to_bf16<<<W_ELEMS / 256, 256, 0, stream>>>(Wo, wob, W_ELEMS);

  // 2) QKV projections (2048 waves each -> 256 blocks of 8 waves)
  //    Q scale = 1/sqrt(64) * log2(e) so attention softmax can use exp2
  k_proj_qkv<<<256, 256, 0, stream>>>(xb, wqb, bq, qbf, 0, 0.125f * 1.44269504f);
  k_proj_qkv<<<256, 256, 0, stream>>>(xb, wkb, bk, kbf, 1, 1.0f);
  k_proj_qkv<<<256, 256, 0, stream>>>(xb, wvb, bv, vtb, 2, 1.0f);

  // 3) attention: 2048 waves = 256 blocks (8 query tiles of one (b,h) each)
  k_attn<<<256, 256, 0, stream>>>(qbf, kbf, vtb, ctx);

  // 4) output projection -> fp32 d_out
  k_out_proj<<<256, 256, 0, stream>>>(ctx, wob, bo, out);
}
